// Block_40415642255965
// MI455X (gfx1250) — matrix operand order repaired and checked by reading, not compiled, not run on hardware
//
#include <hip/hip_runtime.h>
#include <hip/hip_bf16.h>
#include <math.h>

#define Bz 64
#define Nz 197
#define Cz 768
#define Hz 12
#define HDz 64
#define AHz 8
#define AHDz 96
#define AGz 49
#define MTOK (Bz * Nz) /* 12608, divisible by 64 */

typedef _Float16 f16;
typedef __attribute__((ext_vector_type(16))) _Float16 v16h;
typedef __attribute__((ext_vector_type(8)))  _Float16 h8;
typedef __attribute__((ext_vector_type(8)))  float    v8f;
typedef __attribute__((ext_vector_type(4)))  unsigned int u32x4;
typedef __attribute__((ext_vector_type(8)))  int i32x8;
typedef __attribute__((ext_vector_type(4)))  int i32x4;

union HU { v16h v; h8 h[2]; };

// ---------------------------------------------------------------------------
// TDM: DMA a [tileRows x 32] f16 tile (row stride K elements) from global to
// LDS at byte offset lds_off, padding each 32-f16 row (16 DWORDs) with 4
// DWORDs -> LDS row stride 40 f16 (bank-conflict-free for fragment reads).
// D# packing per CDNA5 ISA §8.3-8.4 (group0 128b, group1 256b, 2D tensor).
// ---------------------------------------------------------------------------
__device__ __forceinline__ void tdm_load_tile(unsigned lds_off,
                                              const f16* gptr, int K,
                                              int rows, int tileRows) {
  unsigned long long ga = (unsigned long long)(size_t)gptr;
  u32x4 g0;
  g0.x = 1u;                                   // count=1, user mode, no gather
  g0.y = lds_off;                              // lds_addr (bytes)
  g0.z = (unsigned)ga;                         // global_addr[31:0]
  g0.w = ((unsigned)(ga >> 32) & 0x01FFFFFFu)  // global_addr[56:32]
         | (2u << 30);                         // type=2 ("image")
  i32x8 g1;
  g1[0] = (int)((1u << 16)      // data_size = 1 -> 2 bytes
                | (1u << 20)    // pad_enable
                | (3u << 22)    // pad_interval: 16 DWORDs
                | (3u << 25));  // pad_amount: 4 DWORDs
  g1[1] = (int)((K & 0xFFFF) << 16);                            // dim0[15:0]
  g1[2] = (int)(((K >> 16) & 0xFFFF) | ((rows & 0xFFFF) << 16)); // dim0 hi|dim1 lo
  g1[3] = (int)(((rows >> 16) & 0xFFFF) | (32u << 16));          // dim1 hi|tile_dim0=32
  g1[4] = (int)(tileRows & 0xFFFF);                              // tile_dim1
  g1[5] = K;                                                     // dim0_stride[31:0]
  g1[6] = 0;
  g1[7] = 0;
  i32x4 gz = {0, 0, 0, 0};
#if defined(__clang_major__) && __clang_major__ >= 23
  i32x8 gz8 = {0, 0, 0, 0, 0, 0, 0, 0};
  __builtin_amdgcn_tensor_load_to_lds(g0, g1, gz, gz, gz8, 0);
#else
  __builtin_amdgcn_tensor_load_to_lds(g0, g1, gz, gz, 0);
#endif
}

// ---------------------------------------------------------------------------
// WMMA GEMM:  C[M,Nout] = A[M,K] @ W[Nout,K]^T + bias   (A,W f16; C f32/f16)
// Block: 256 threads = 8 waves; block tile 64(M) x 128(N); wave tile 32x32
// (2x2 WMMA tiles). A/B staged into double-buffered LDS by the Tensor Data
// Mover (wave 0 issues, s_wait_tensorcnt + barrier synchronizes).
// ACT: 0 = none, 1 = exact GELU.
// ---------------------------------------------------------------------------
template <int ACT>
__global__ __launch_bounds__(256) void gemm_wmma(
    const f16* __restrict__ A, const f16* __restrict__ W,
    const float* __restrict__ bias,
    float* __restrict__ Cf, f16* __restrict__ Ch,
    int K, int Nout) {
  __shared__ __align__(16) f16 sA[2][64][40];   // 2 x 5120 B
  __shared__ __align__(16) f16 sB[2][128][40];  // 2 x 10240 B

  const int tid  = threadIdx.x;
  const int lane = tid & 31;
  const int wid  = tid >> 5;
  const int wm   = wid & 1;   // wave M group (0..1): rows wm*32..wm*32+31
  const int wn   = wid >> 1;  // wave N group (0..3): cols wn*32..wn*32+31

  const int m0 = blockIdx.y * 64;
  const int n0 = blockIdx.x * 128;

  const f16* Abase = A + (size_t)m0 * K;
  const f16* Bbase = W + (size_t)n0 * K;

  v8f acc[2][2] = {};

  const int kh = (lane >> 4) * 8;   // A K-half
  const int bk = (lane >> 4) * 8;   // B K-half
  const int l15 = lane & 15;

  const int nsteps = K / 32;
  if (wid == 0) {  // TDM prologue: fill buffer 0
    tdm_load_tile((unsigned)(size_t)&sA[0][0][0], Abase, K, 64, 64);
    tdm_load_tile((unsigned)(size_t)&sB[0][0][0], Bbase, K, 128, 128);
  }

  for (int i = 0; i < nsteps; ++i) {
    const int cur = i & 1;
    if (wid == 0) {
      if (i + 1 < nsteps) {  // prefetch next K-step into other buffer
        tdm_load_tile((unsigned)(size_t)&sA[cur ^ 1][0][0],
                      Abase + (size_t)(i + 1) * 32, K, 64, 64);
        tdm_load_tile((unsigned)(size_t)&sB[cur ^ 1][0][0],
                      Bbase + (size_t)(i + 1) * 32, K, 128, 128);
        __builtin_amdgcn_s_wait_tensorcnt(2);  // in-order: current buf ready
      } else {
        __builtin_amdgcn_s_wait_tensorcnt(0);
      }
    }
    __syncthreads();

    HU af[2], bf[2];
    for (int mt = 0; mt < 2; ++mt) {
      int mrow = wm * 32 + mt * 16 + l15;
      af[mt].h[0] = *(const h8*)&sA[cur][mrow][kh];
      af[mt].h[1] = *(const h8*)&sA[cur][mrow][kh + 16];
    }
    for (int nt = 0; nt < 2; ++nt) {
      int ncol = wn * 32 + nt * 16 + l15;
      bf[nt].h[0] = *(const h8*)&sB[cur][ncol][bk];
      bf[nt].h[1] = *(const h8*)&sB[cur][ncol][bk + 16];
    }
    for (int mt = 0; mt < 2; ++mt)
      for (int nt = 0; nt < 2; ++nt)
        acc[mt][nt] = __builtin_amdgcn_wmma_f32_16x16x32_f16(
            false, af[mt].v, false, bf[nt].v, (short)0, acc[mt][nt], false,
            false);
    __syncthreads();  // all waves done reading before buf is overwritten
  }

  // Epilogue: D layout -> lane holds col n=(lane&15), rows r2 + (lane>>4)*8
  for (int mt = 0; mt < 2; ++mt) {
    int mb = m0 + wm * 32 + mt * 16 + (lane >> 4) * 8;
    for (int nt = 0; nt < 2; ++nt) {
      int gn = n0 + wn * 32 + nt * 16 + l15;
      float bvv = bias ? bias[gn] : 0.f;
      for (int r2 = 0; r2 < 8; ++r2) {
        float val = acc[mt][nt][r2] + bvv;
        if (ACT == 1) val = 0.5f * val * (1.f + erff(val * 0.70710678118f));
        size_t off = (size_t)(mb + r2) * Nout + gn;
        if (Cf) Cf[off] = val;
        if (Ch) Ch[off] = (f16)val;
      }
    }
  }
}

// ---------------------------------------------------------------------------
// LayerNorm over C=768, one wave per token. Optional f32 / f16 outputs.
// ---------------------------------------------------------------------------
__global__ __launch_bounds__(256) void ln_kernel(
    const float* __restrict__ in, const float* __restrict__ w,
    const float* __restrict__ b, float* __restrict__ out,
    f16* __restrict__ out16, int ntok) {
  int gw = blockIdx.x * 8 + (threadIdx.x >> 5);
  if (gw >= ntok) return;
  int lane = threadIdx.x & 31;
  const float* row = in + (size_t)gw * Cz;
  float s = 0.f, ss = 0.f;
  for (int i = lane; i < Cz; i += 32) { float v = row[i]; s += v; ss += v * v; }
  for (int o = 16; o > 0; o >>= 1) {
    s += __shfl_xor(s, o, 32);
    ss += __shfl_xor(ss, o, 32);
  }
  float mean = s * (1.f / Cz);
  float var  = ss * (1.f / Cz) - mean * mean;
  float rstd = rsqrtf(var + 1e-5f);
  for (int i = lane; i < Cz; i += 32) {
    float v = (row[i] - mean) * rstd * w[i] + b[i];
    if (out)   out[(size_t)gw * Cz + i] = v;
    if (out16) out16[(size_t)gw * Cz + i] = (f16)v;
  }
}

// ---------------------------------------------------------------------------
// Generic softmax attention, one wave per (b, h, query-row).
// Q/K/V/O addressed as base[(b*L + tok)*stride + h*hd + d].
// ---------------------------------------------------------------------------
__global__ __launch_bounds__(256) void attn_kernel(
    const float* __restrict__ Q, int qs, const float* __restrict__ Km, int ks,
    const float* __restrict__ Vm, int vs, float* __restrict__ O, int os,
    f16* __restrict__ O16, int Lq, int Lk, int hd, int Hh, float scale) {
  __shared__ float s_q[8][96];
  __shared__ float s_sc[8][208];
  int gw = blockIdx.x * 8 + (threadIdx.x >> 5);
  int total = Bz * Hh * Lq;
  if (gw >= total) return;
  int lane = threadIdx.x & 31;
  int wid  = threadIdx.x >> 5;
  int iq = gw % Lq;
  int h  = (gw / Lq) % Hh;
  int b  = gw / (Lq * Hh);

  const float* qr = Q + (size_t)(b * Lq + iq) * qs + h * hd;
  for (int d = lane; d < hd; d += 32) s_q[wid][d] = qr[d] * scale;

  float mx = -3.4e38f;
  for (int n = lane; n < Lk; n += 32) {
    const float* kr = Km + (size_t)(b * Lk + n) * ks + h * hd;
    float s = 0.f;
    for (int d = 0; d < hd; d += 4) {
      float4 kk = *(const float4*)(kr + d);
      s += s_q[wid][d] * kk.x + s_q[wid][d + 1] * kk.y +
           s_q[wid][d + 2] * kk.z + s_q[wid][d + 3] * kk.w;
    }
    s_sc[wid][n] = s;
    mx = fmaxf(mx, s);
  }
  for (int o = 16; o > 0; o >>= 1) mx = fmaxf(mx, __shfl_xor(mx, o, 32));
  float sum = 0.f;
  for (int n = lane; n < Lk; n += 32) {
    float e = expf(s_sc[wid][n] - mx);
    s_sc[wid][n] = e;
    sum += e;
  }
  for (int o = 16; o > 0; o >>= 1) sum += __shfl_xor(sum, o, 32);
  float inv = 1.f / sum;

  for (int d = lane; d < hd; d += 32) {
    float acc = 0.f;
    for (int n = 0; n < Lk; ++n)
      acc += s_sc[wid][n] * Vm[(size_t)(b * Lk + n) * vs + h * hd + d];
    float val = acc * inv;
    size_t off = (size_t)(b * Lq + iq) * os + h * hd + d;
    O[off] = val;
    if (O16) O16[off] = (f16)val;
  }
}

// ---------------------------------------------------------------------------
// Depthwise conv1d over tokens, k=3 pad=1. addFlag: accumulate into out.
// ---------------------------------------------------------------------------
__global__ __launch_bounds__(256) void dwconv_kernel(
    const float* __restrict__ in, const float* __restrict__ w,
    const float* __restrict__ bias, float* __restrict__ out,
    f16* __restrict__ out16, int L, int ibs, int its, int obs, int ots,
    int addFlag) {
  size_t idx = (size_t)blockIdx.x * 256 + threadIdx.x;
  size_t total = (size_t)Bz * L * Cz;
  if (idx >= total) return;
  int c = idx % Cz;
  int n = (idx / Cz) % L;
  int b = idx / ((size_t)Cz * L);
  float acc = bias[c];
  for (int j = 0; j < 3; ++j) {
    int t = n - 1 + j;
    if (t >= 0 && t < L)
      acc += w[c * 3 + j] * in[(size_t)b * ibs + (size_t)t * its + c];
  }
  size_t oo = (size_t)b * obs + (size_t)n * ots + c;
  if (addFlag) out[oo] += acc;
  else out[oo] = acc;
  if (out16) out16[oo] = (f16)(addFlag ? out[oo] : acc);
}

// agent[b,a,c] = mean over 4 tokens of q slice in qkv (B,N,3C)
__global__ __launch_bounds__(256) void pool_kernel(
    const float* __restrict__ qkv, float* __restrict__ agent) {
  size_t idx = (size_t)blockIdx.x * 256 + threadIdx.x;
  size_t total = (size_t)Bz * AGz * Cz;
  if (idx >= total) return;
  int c = idx % Cz;
  int a = (idx / Cz) % AGz;
  int b = idx / ((size_t)Cz * AGz);
  const float* base = qkv + ((size_t)b * Nz + a * 4) * (3 * Cz) + c;
  float s = 0.f;
  for (int j = 0; j < 4; ++j) s += base[(size_t)j * 3 * Cz];
  agent[((size_t)b * AGz + a) * Cz + c] = s * 0.25f;
}

__global__ __launch_bounds__(256) void add2_kernel(
    const float* __restrict__ a, const float* __restrict__ b,
    float* __restrict__ o, f16* __restrict__ o16, size_t n) {
  size_t i = (size_t)blockIdx.x * 256 + threadIdx.x;
  if (i >= n) return;
  float v = a[i] + b[i];
  o[i] = v;
  if (o16) o16[i] = (f16)v;
}

__global__ __launch_bounds__(256) void cvt16_kernel(
    const float* __restrict__ a, f16* __restrict__ o, size_t n) {
  size_t i = (size_t)blockIdx.x * 256 + threadIdx.x;
  if (i < n) o[i] = (f16)a[i];
}

// ---------------------------------------------------------------------------
extern "C" void kernel_launch(void* const* d_in, const int* in_sizes, int n_in,
                              void* d_out, int out_size, void* d_ws,
                              size_t ws_size, hipStream_t stream) {
  const float* x        = (const float*)d_in[0];
  const float* n1w      = (const float*)d_in[1];
  const float* n1b      = (const float*)d_in[2];
  const float* qkv_w    = (const float*)d_in[3];
  const float* qkv_b    = (const float*)d_in[4];
  const float* proj_w   = (const float*)d_in[5];
  const float* proj_b   = (const float*)d_in[6];
  const float* dwc_w    = (const float*)d_in[7];
  const float* dwc_b    = (const float*)d_in[8];
  const float* normy_w  = (const float*)d_in[9];
  const float* normy_b  = (const float*)d_in[10];
  const float* anw      = (const float*)d_in[11];
  const float* anb      = (const float*)d_in[12];
  const float* in0_w    = (const float*)d_in[13];
  const float* in0_b    = (const float*)d_in[14];
  const float* out0_w   = (const float*)d_in[15];
  const float* out0_b   = (const float*)d_in[16];
  const float* ln0_w    = (const float*)d_in[17];
  const float* ln0_b    = (const float*)d_in[18];
  const float* in1_w    = (const float*)d_in[19];
  const float* in1_b    = (const float*)d_in[20];
  const float* out1_w   = (const float*)d_in[21];
  const float* out1_b   = (const float*)d_in[22];
  const float* n2w      = (const float*)d_in[23];
  const float* n2b      = (const float*)d_in[24];
  const float* fc1_w    = (const float*)d_in[25];
  const float* fc1_b    = (const float*)d_in[26];
  const float* fc2_w    = (const float*)d_in[27];
  const float* fc2_b    = (const float*)d_in[28];
  float* outp = (float*)d_out;

  const size_t MC = (size_t)MTOK * Cz;
  const size_t AC = (size_t)Bz * AGz * Cz;
  const size_t CC = (size_t)Cz * Cz;

  char* p = (char*)d_ws;
  auto alloc = [&](size_t bytes) {
    void* r = (void*)p;
    p += (bytes + 255) & ~(size_t)255;
    return r;
  };
  float* h      = (float*)alloc(MC * 4);
  float* qkv    = (float*)alloc(3 * MC * 4);  // later: qm | kvm, then mlp-out
  float* y      = (float*)alloc(MC * 4);      // y / attn-out / proj-out
  float* tgt    = (float*)alloc(MC * 4);
  float* mem    = (float*)alloc(MC * 4);
  float* x1     = (float*)alloc(MC * 4);
  float* agent  = (float*)alloc(AC * 4);
  float* agentv = (float*)alloc(AC * 4);
  f16* h16   = (f16*)alloc(MC * 2);
  f16* tgt16 = (f16*)alloc(MC * 2);
  f16* mem16 = (f16*)alloc(MC * 2);
  f16* ao16  = (f16*)alloc(MC * 2);
  f16* act16 = (f16*)alloc(4 * MC * 2);
  f16* wqkv16 = (f16*)alloc(3 * CC * 2);
  f16* wprj16 = (f16*)alloc(CC * 2);
  f16* win0   = (f16*)alloc(3 * CC * 2);
  f16* wou0   = (f16*)alloc(CC * 2);
  f16* win1   = (f16*)alloc(3 * CC * 2);
  f16* wou1   = (f16*)alloc(CC * 2);
  f16* wfc1   = (f16*)alloc(4 * CC * 2);
  f16* wfc2   = (f16*)alloc(4 * CC * 2);

  float* qm  = qkv;        // MTOK x C
  float* kvm = qkv + MC;   // MTOK x 2C
  float* mlp = qkv;        // MTOK x C (after MHA done)

  auto cvt = [&](const float* a, f16* o, size_t n) {
    cvt16_kernel<<<(int)((n + 255) / 256), 256, 0, stream>>>(a, o, n);
  };
  auto gemm = [&](const f16* A, const f16* W, const float* bias, float* Cf,
                  f16* Ch, int K, int Nout, int act) {
    dim3 g(Nout / 128, MTOK / 64);
    if (act) gemm_wmma<1><<<g, 256, 0, stream>>>(A, W, bias, Cf, Ch, K, Nout);
    else     gemm_wmma<0><<<g, 256, 0, stream>>>(A, W, bias, Cf, Ch, K, Nout);
  };
  auto attn = [&](const float* Q, int qs, const float* K, int ks,
                  const float* V, int vs, float* O, int os, f16* O16, int Lq,
                  int Lk, int hd, int Hh, float scale) {
    int waves = Bz * Hh * Lq;
    attn_kernel<<<(waves + 7) / 8, 256, 0, stream>>>(Q, qs, K, ks, V, vs, O,
                                                     os, O16, Lq, Lk, hd, Hh,
                                                     scale);
  };
  auto ln = [&](const float* in, const float* w, const float* b, float* o,
                f16* o16) {
    ln_kernel<<<(MTOK + 7) / 8, 256, 0, stream>>>(in, w, b, o, o16, MTOK);
  };
  auto add2 = [&](const float* a, const float* b, float* o, f16* o16,
                  size_t n) {
    add2_kernel<<<(int)((n + 255) / 256), 256, 0, stream>>>(a, b, o, o16, n);
  };

  // weight conversions (f32 -> f16)
  cvt(qkv_w, wqkv16, 3 * CC);  cvt(proj_w, wprj16, CC);
  cvt(in0_w, win0, 3 * CC);    cvt(out0_w, wou0, CC);
  cvt(in1_w, win1, 3 * CC);    cvt(out1_w, wou1, CC);
  cvt(fc1_w, wfc1, 4 * CC);    cvt(fc2_w, wfc2, 4 * CC);

  // ---- AgentAttention ----
  ln(x, n1w, n1b, h, h16);                                  // h = LN(x)
  gemm(h16, wqkv16, qkv_b, qkv, nullptr, Cz, 3 * Cz, 0);    // qkv
  pool_kernel<<<(int)((AC + 255) / 256), 256, 0, stream>>>(qkv, agent);
  const float scA = 0.125f;  // HD^-0.5
  attn(agent, Cz, qkv + Cz, 3 * Cz, qkv + 2 * Cz, 3 * Cz, agentv, Cz, nullptr,
       AGz, Nz, HDz, Hz, scA);                               // agent_v
  attn(qkv, 3 * Cz, agent, Cz, agentv, Cz, y, Cz, nullptr, Nz, AGz, HDz, Hz,
       scA);                                                 // y
  {  // y[:, :-1] += dwconv(v[:, :-1])
    size_t total = (size_t)Bz * (Nz - 1) * Cz;
    dwconv_kernel<<<(int)((total + 255) / 256), 256, 0, stream>>>(
        qkv + 2 * Cz, dwc_w, dwc_b, y, nullptr, Nz - 1, Nz * 3 * Cz, 3 * Cz,
        Nz * Cz, Cz, 1);
  }

  // ---- CA fusion branch ----
  ln(y, anw, anb, tgt, tgt16);  // tgt0
  {  // mem = dwconv(h)
    size_t total = (size_t)Bz * Nz * Cz;
    dwconv_kernel<<<(int)((total + 255) / 256), 256, 0, stream>>>(
        h, normy_w, normy_b, mem, mem16, Nz, Nz * Cz, Cz, Nz * Cz, Cz, 0);
  }
  const float scM = 0.1020620726f;  // (C/AH)^-0.5 = 96^-0.5
  // mha0
  gemm(tgt16, win0, in0_b, qm, nullptr, Cz, Cz, 0);
  gemm(mem16, win0 + CC, in0_b + Cz, kvm, nullptr, Cz, 2 * Cz, 0);
  attn(qm, Cz, kvm, 2 * Cz, kvm + Cz, 2 * Cz, y, Cz, ao16, Nz, Nz, AHDz, AHz,
       scM);
  gemm(ao16, wou0, out0_b, qm, nullptr, Cz, Cz, 0);
  add2(tgt, qm, tgt, nullptr, MC);         // tgt += mha0
  ln(tgt, ln0_w, ln0_b, tgt, tgt16);       // in-place LN
  // mha1
  gemm(tgt16, win1, in1_b, qm, nullptr, Cz, Cz, 0);
  gemm(mem16, win1 + CC, in1_b + Cz, kvm, nullptr, Cz, 2 * Cz, 0);
  attn(qm, Cz, kvm, 2 * Cz, kvm + Cz, 2 * Cz, y, Cz, ao16, Nz, Nz, AHDz, AHz,
       scM);
  gemm(ao16, wou1, out1_b, qm, nullptr, Cz, Cz, 0);
  add2(tgt, qm, tgt, nullptr, MC);         // tgt += mha1
  add2(h, tgt, tgt, tgt16, MC);            // y = h + tgt
  gemm(tgt16, wprj16, proj_b, y, nullptr, Cz, Cz, 0);  // y @ proj^T + b

  // ---- residual + MLP ----
  add2(x, y, x1, nullptr, MC);             // x1 = x + y
  ln(x1, n2w, n2b, nullptr, h16);          // m (f16 only)
  gemm(h16, wfc1, fc1_b, nullptr, act16, Cz, 4 * Cz, 1);  // GELU(fc1)
  gemm(act16, wfc2, fc2_b, mlp, nullptr, 4 * Cz, Cz, 0);  // fc2
  add2(x1, mlp, outp, nullptr, MC);        // out = x1 + mlp
}
